// CausalSelfAttention_21509196218626
// MI455X (gfx1250) — matrix operand order repaired and checked by reading, not compiled, not run on hardware
//
#include <hip/hip_runtime.h>
#include <hip/hip_bf16.h>

// Causal self-attention for MI455X (gfx1250, wave32, WMMA).
// S=2048, D=1024, H=16, HS=64.  All GEMMs + attention run through
// v_wmma_f32_16x16x32_f16 with f32 accumulation; GEMM B-panels are staged
// into LDS with CDNA5 async global->LDS copies (ASYNCcnt) double-buffered.

#define S_LEN 2048
#define D_DIM 1024
#define NH    16
#define HS    64
#define TRIPLE_D (3 * D_DIM)

typedef _Float16 v16h __attribute__((ext_vector_type(16)));
typedef float    v8f  __attribute__((ext_vector_type(8)));
typedef int      v4i  __attribute__((ext_vector_type(4)));
typedef __attribute__((address_space(1))) v4i gas_v4i;   // global
typedef __attribute__((address_space(3))) v4i las_v4i;   // LDS

union FragU { v16h v; uint4 q[2]; };

// A-fragment (16x32 f16): lane half 0 -> K 0..15,
// lane half 1 -> K 16..31. Caller passes p = &row[k0 + half*16].
__device__ __forceinline__ v16h ld_frag_a(const _Float16* p) {
    FragU f;
    f.q[0] = *(const uint4*)(p);
    f.q[1] = *(const uint4*)(p + 8);
    return f.v;
}
// B-fragment (32x16 f16): lane owns column N = lane&15; half 0 -> K 0..15,
// half 1 -> K 16..31, contiguous when B is stored as [N][K] row-major.
__device__ __forceinline__ v16h ld_frag_b(const _Float16* p) {
    FragU f;
    f.q[0] = *(const uint4*)(p);
    f.q[1] = *(const uint4*)(p + 8);
    return f.v;
}

__device__ __forceinline__ v8f wmma16(v16h a, v16h b, v8f c) {
    return __builtin_amdgcn_wmma_f32_16x16x32_f16(false, a, false, b,
                                                  (short)0, c, false, false);
}

// ---- CDNA5 async global->LDS 16-byte copy (ASYNCcnt-tracked) ----
__device__ __forceinline__ void stage16(const _Float16* g, _Float16* l) {
#if __has_builtin(__builtin_amdgcn_global_load_async_to_lds_b128)
    __builtin_amdgcn_global_load_async_to_lds_b128(
        (gas_v4i*)g, (las_v4i*)l, 0, 0);
#else
    unsigned lofs = (unsigned)(size_t)l;           // LDS byte offset
    unsigned long long ga = (unsigned long long)(size_t)g;
    asm volatile("global_load_async_to_lds_b128 %0, %1, off"
                 :: "v"(lofs), "v"(ga) : "memory");
#endif
}
__device__ __forceinline__ void wait_async0() {
#if __has_builtin(__builtin_amdgcn_s_wait_asynccnt)
    __builtin_amdgcn_s_wait_asynccnt(0);
#else
    asm volatile("s_wait_asynccnt 0x0" ::: "memory");
#endif
}

// ---------------- precision-conversion kernels ----------------

__global__ void cvt_f32_to_f16(const float* __restrict__ in,
                               _Float16* __restrict__ out, int n) {
    int i = blockIdx.x * 256 + threadIdx.x;
    if (i < n) out[i] = (_Float16)in[i];
}

// in: [K,N] f32 row-major  ->  out: [N,K] f16 row-major (transpose+convert)
__global__ void cvt_transpose_f16(const float* __restrict__ in,
                                  _Float16* __restrict__ out, int K, int N) {
    int i = blockIdx.x * 256 + threadIdx.x;
    if (i < K * N) {
        int k = i / N;
        int n = i - k * N;
        out[(size_t)n * K + k] = (_Float16)in[i];
    }
}

// ---------------- unified WMMA GEMM ----------------
// Block = 256 thr (8 waves). Block tile: M=256 (wave owns 32 rows), N=64.
// B K-panel (64 n-rows x 32 k) staged to LDS via async copies, double
// buffered; rows padded to 72 halfs (144B) for conflict-free ds_load_b128.
// EPI=0: out0 = f32 [M,D] dense (+bias).  EPI=1: QKV routing to out0/1/2.

#define BPAD 72

template <int EPI>
__global__ __launch_bounds__(256) void gemm_wmma(
        const _Float16* __restrict__ A,    // [M, 1024]
        const _Float16* __restrict__ Bt,   // [Ntot, 1024]
        const float*    __restrict__ bias, // [Ntot]
        void* out0, void* out1, void* out2) {
    __shared__ __align__(16) _Float16 bpanel[2][64 * BPAD];

    const int tid  = threadIdx.x;
    const int lane = tid & 31;
    const int wave = tid >> 5;
    const int r16  = lane & 15;
    const int half = lane >> 4;
    const int mBase = blockIdx.x * 256 + wave * 32;
    const int nBase = blockIdx.y * 64;

    // staging slice for this thread: 16B of row sn at k-chunk sk
    const int sn = tid >> 2;            // 0..63
    const int sk = (tid & 3) * 8;       // halfs
    const _Float16* gsrc = Bt + (size_t)(nBase + sn) * D_DIM + sk;
    _Float16* lds0 = &bpanel[0][sn * BPAD + sk];
    _Float16* lds1 = &bpanel[1][sn * BPAD + sk];

    stage16(gsrc, lds0);                // panel for k0 = 0

    v8f acc[2][4] = {};
    const _Float16* arow0 = A + (size_t)(mBase + r16) * D_DIM + half * 16;
    const _Float16* arow1 = arow0 + (size_t)16 * D_DIM;

    v16h a0 = ld_frag_a(arow0);
    v16h a1 = ld_frag_a(arow1);

    const int NK = D_DIM / 32;
    for (int i = 0; i < NK; ++i) {
        wait_async0();                  // my panel-i copy landed
        __syncthreads();                // everyone's panel-i landed; panel
                                        // (i+1)&1 free to overwrite
        if (i + 1 < NK)
            stage16(gsrc + (i + 1) * 32, (i & 1) ? lds0 : lds1);
        // prefetch next A fragments before consuming current ones
        v16h a0n = a0, a1n = a1;
        if (i + 1 < NK) {
            a0n = ld_frag_a(arow0 + (i + 1) * 32);
            a1n = ld_frag_a(arow1 + (i + 1) * 32);
        }
        const _Float16* lb = &bpanel[i & 1][0];
#pragma unroll
        for (int t = 0; t < 4; ++t) {
            v16h b = ld_frag_b(lb + (t * 16 + r16) * BPAD + half * 16);
            acc[0][t] = wmma16(a0, b, acc[0][t]);
            acc[1][t] = wmma16(a1, b, acc[1][t]);
        }
        a0 = a0n; a1 = a1n;
    }

    if (EPI == 0) {
        float* out = (float*)out0;
#pragma unroll
        for (int w = 0; w < 2; ++w) {
#pragma unroll
            for (int t = 0; t < 4; ++t) {
                const int col = nBase + t * 16 + r16;
                const float bv = bias[col];
#pragma unroll
                for (int r = 0; r < 8; ++r) {
                    const int row = mBase + w * 16 + (half << 3) + r;
                    out[(size_t)row * D_DIM + col] = acc[w][t][r] + bv;
                }
            }
        }
    } else {
        _Float16* Qb = (_Float16*)out0;
        _Float16* Kb = (_Float16*)out1;
        _Float16* Vt = (_Float16*)out2;
#pragma unroll
        for (int w = 0; w < 2; ++w) {
#pragma unroll
            for (int t = 0; t < 4; ++t) {
                const int col = nBase + t * 16 + r16;
                const float bv = bias[col];
                const int sec = col >> 10;        // 0=Q 1=K 2=V
                const int cc  = col & (D_DIM - 1);
                const int h   = cc >> 6;
                const int hs  = cc & (HS - 1);
#pragma unroll
                for (int r = 0; r < 8; ++r) {
                    const int row = mBase + w * 16 + (half << 3) + r;
                    const _Float16 val = (_Float16)(acc[w][t][r] + bv);
                    if (sec == 0)      Qb[((size_t)h * S_LEN + row) * HS + hs] = val;
                    else if (sec == 1) Kb[((size_t)h * S_LEN + row) * HS + hs] = val;
                    else               Vt[((size_t)h * HS + hs) * S_LEN + row] = val;
                }
            }
        }
    }
}

// ---------------- flash attention (per head, 16-query tiles) ----------------
// grid = (H, S/16), block = 32 (one wave) -> 2048 single-wave workgroups for
// fine-grained load balancing of the causal triangle.  Key blocks of 32;
// online softmax; P staged through per-wave LDS (C-layout -> A-layout).
// Q fragments are re-read from cache each key block instead of being kept
// live across the loop (avoids scratch spills of WMMA operands).

__global__ __launch_bounds__(32) void attn_f16(
        const _Float16* __restrict__ Qb,
        const _Float16* __restrict__ Kb,
        const _Float16* __restrict__ Vt,
        _Float16* __restrict__ Ob) {          // [S, D]
    __shared__ __align__(16) _Float16 ldsW[16 * 32];
    const int lane = threadIdx.x & 31;
    const int r16  = lane & 15;
    const int half = lane >> 4;
    const int h  = blockIdx.x;
    const int qb = blockIdx.y * 16;

    const _Float16* Qh = Qb + (size_t)h * S_LEN * HS;
    const _Float16* Kh = Kb + (size_t)h * S_LEN * HS;
    const _Float16* Vh = Vt + (size_t)h * HS * S_LEN;

    const _Float16* qrow = Qh + (size_t)(qb + r16) * HS + half * 16;

    v8f o[4] = {};
    float m[8], l[8];
#pragma unroll
    for (int r = 0; r < 8; ++r) { m[r] = -3.0e38f; l[r] = 0.0f; }

    const float scale = 0.02209708691f;   // 1/sqrt(2048)
    const int ktmax = (qb + 15) >> 5;

    for (int kt = 0; kt <= ktmax; ++kt) {
        const int kb = kt * 32;
        // scores: two 16x16 column tiles, K=64 contraction (2 wmma each).
        // Q fragments re-read here (hits L0; short live range, no spills).
        v8f s0 = {}, s1 = {};
        {
            const v16h aq0 = ld_frag_a(qrow);
            const v16h aq1 = ld_frag_a(qrow + 32);
            const _Float16* k0p = Kh + (size_t)(kb + r16) * HS + half * 16;
            const _Float16* k1p = Kh + (size_t)(kb + 16 + r16) * HS + half * 16;
            s0 = wmma16(aq0, ld_frag_b(k0p), s0);
            s0 = wmma16(aq1, ld_frag_b(k0p + 32), s0);
            s1 = wmma16(aq0, ld_frag_b(k1p), s1);
            s1 = wmma16(aq1, ld_frag_b(k1p + 32), s1);
        }
        const bool needMask = (kb + 31 > qb);
#pragma unroll
        for (int r = 0; r < 8; ++r) {
            float s0r = s0[r] * scale;
            float s1r = s1[r] * scale;
            if (needMask) {
                const int qi = qb + (half << 3) + r;
                if (kb + r16 > qi)      s0r = -1.0e30f;
                if (kb + 16 + r16 > qi) s1r = -1.0e30f;
            }
            // row-max over the 16 lanes of this half-wave
            float rm = fmaxf(s0r, s1r);
#pragma unroll
            for (int x = 1; x < 16; x <<= 1) rm = fmaxf(rm, __shfl_xor(rm, x, 32));
            const float mnew  = fmaxf(m[r], rm);
            const float alpha = __expf(m[r] - mnew);
            const float p0 = __expf(s0r - mnew);
            const float p1 = __expf(s1r - mnew);
            float rs = p0 + p1;
#pragma unroll
            for (int x = 1; x < 16; x <<= 1) rs += __shfl_xor(rs, x, 32);
            l[r] = l[r] * alpha + rs;
            m[r] = mnew;
#pragma unroll
            for (int t = 0; t < 4; ++t) o[t][r] *= alpha;
            // stage P (row-major 16x32) for A-fragment reload
            const int prow = (half << 3) + r;
            ldsW[prow * 32 + r16]      = (_Float16)p0;
            ldsW[prow * 32 + 16 + r16] = (_Float16)p1;
        }
        // same-wave DS ordering; make the re-load wait on the stores
        asm volatile("s_wait_dscnt 0" ::: "memory");
        const v16h pa = ld_frag_a(ldsW + r16 * 32 + half * 16);
#pragma unroll
        for (int t = 0; t < 4; ++t) {
            v16h bv = ld_frag_b(Vh + (size_t)(t * 16 + r16) * S_LEN
                                + kb + half * 16);
            o[t] = wmma16(pa, bv, o[t]);
        }
    }
    // normalize + merge heads -> Ob[s, h*64 + d]
#pragma unroll
    for (int r = 0; r < 8; ++r) {
        const float inv = 1.0f / l[r];
        const int row = qb + (half << 3) + r;
#pragma unroll
        for (int t = 0; t < 4; ++t)
            Ob[(size_t)row * D_DIM + h * HS + t * 16 + r16] =
                (_Float16)(o[t][r] * inv);
    }
}

// ---------------- launch ----------------

extern "C" void kernel_launch(void* const* d_in, const int* in_sizes, int n_in,
                              void* d_out, int out_size, void* d_ws, size_t ws_size,
                              hipStream_t stream) {
    const float* hs     = (const float*)d_in[0];   // [2048,1024]
    const float* W_attn = (const float*)d_in[1];   // [1024,3072]
    const float* b_attn = (const float*)d_in[2];   // [3072]
    const float* W_proj = (const float*)d_in[3];   // [1024,1024]
    const float* b_proj = (const float*)d_in[4];   // [1024]
    float* out = (float*)d_out;

    _Float16* Xh  = (_Float16*)d_ws;                     // 2048*1024
    _Float16* WaT = Xh  + (size_t)S_LEN * D_DIM;         // 3072*1024
    _Float16* WpT = WaT + (size_t)TRIPLE_D * D_DIM;      // 1024*1024
    _Float16* Qb  = WpT + (size_t)D_DIM * D_DIM;         // 16*2048*64
    _Float16* Kb  = Qb  + (size_t)NH * S_LEN * HS;
    _Float16* Vt  = Kb  + (size_t)NH * S_LEN * HS;
    _Float16* Ob  = Vt  + (size_t)NH * S_LEN * HS;       // 2048*1024

    const int nX  = S_LEN * D_DIM;
    const int nWa = D_DIM * TRIPLE_D;
    const int nWp = D_DIM * D_DIM;

    cvt_f32_to_f16<<<(nX + 255) / 256, 256, 0, stream>>>(hs, Xh, nX);
    cvt_transpose_f16<<<(nWa + 255) / 256, 256, 0, stream>>>(W_attn, WaT, D_DIM, TRIPLE_D);
    cvt_transpose_f16<<<(nWp + 255) / 256, 256, 0, stream>>>(W_proj, WpT, D_DIM, D_DIM);

    gemm_wmma<1><<<dim3(S_LEN / 256, TRIPLE_D / 64), 256, 0, stream>>>(
        Xh, WaT, b_attn, (void*)Qb, (void*)Kb, (void*)Vt);

    attn_f16<<<dim3(NH, S_LEN / 16), 32, 0, stream>>>(Qb, Kb, Vt, Ob);

    gemm_wmma<0><<<dim3(S_LEN / 256, D_DIM / 64), 256, 0, stream>>>(
        Ob, WpT, b_proj, (void*)out, nullptr, nullptr);
}
